// SimpleDecoder_64158221467836
// MI455X (gfx1250) — matrix operand order repaired and checked by reading, not compiled, not run on hardware
//
#include <hip/hip_runtime.h>
#include <hip/hip_bf16.h>

// ---------------------------------------------------------------------------
// SimpleDecoder on gfx1250 (MI455X): bf16 WMMA everywhere, persistent LSTM.
// Dims fixed by the reference: B=8, T=128, S=256, H=1024, V=32000.
// ---------------------------------------------------------------------------
#define DB 8
#define DT 128
#define DSS 256
#define DH 1024
#define DV 32000

typedef __attribute__((ext_vector_type(16))) __bf16        v16bf;
typedef __attribute__((ext_vector_type(8)))  float         v8f;
typedef __attribute__((ext_vector_type(8)))  unsigned short v8us;

union FragU { v16bf v; v8us h[2]; };

__device__ inline v8f vzero() {
  v8f z = {0.f, 0.f, 0.f, 0.f, 0.f, 0.f, 0.f, 0.f};
  return z;
}

// fp32 -> bf16 round-to-nearest-even
__device__ inline unsigned short f2bf(float f) {
  unsigned u = __float_as_uint(f);
  unsigned r = ((u >> 16) & 1u) + 0x7fffu;
  return (unsigned short)((u + r) >> 16);
}

// A fragment: 16x32 bf16 tile, A is K-contiguous row-major, base = row m0.
// Lane L (L<16): M=L, K in {0..7, 16..23}; lane>=16: M=L-16, K in {8..15, 24..31}.
__device__ inline v16bf load_a_frag(const unsigned short* __restrict__ A, int lda, int k0) {
  int lane = threadIdx.x & 31;
  const unsigned short* p = A + (size_t)(lane & 15) * lda + k0 + 8 * (lane >> 4);
  FragU f;
  f.h[0] = *(const v8us*)(p);
  f.h[1] = *(const v8us*)(p + 16);
  return f.v;
}

// B fragment: 32x16 bf16 tile where B[k][n] = W[n][k], W K-contiguous row-major,
// base = row n0. Lane L: N = L&15 (row of W), K in {0..7, 16..23} (L<16) or {8..15, 24..31} (L>=16).
__device__ inline v16bf load_b_frag(const unsigned short* __restrict__ W, int ldw, int k0) {
  int lane = threadIdx.x & 31;
  const unsigned short* p = W + (size_t)(lane & 15) * ldw + k0 + 8 * (lane >> 4);
  FragU f;
  f.h[0] = *(const v8us*)(p);
  f.h[1] = *(const v8us*)(p + 16);
  return f.v;
}

__device__ inline v8f wmma_bf16(v16bf a, v16bf b, v8f c) {
  return __builtin_amdgcn_wmma_f32_16x16x32_bf16(false, a, false, b, (short)0, c, false, false);
}

// ---------------------------------------------------------------------------
// Generic wave-tile GEMM: C[m,n] = scale * sum_k A[m,k]*W[n,k] + bias(+bias2)
// One wave computes a 16x64 C tile (4 accumulators). 8 waves / block.
// K loop uses a 4-deep rotating fragment pipeline (K unrolled by 128): the
// reload of buffer u sits 3 WMMA-groups ahead of its consumer, so the
// scheduler can keep ~30 b128 loads in flight behind the MMA pipe while the
// allocator writes load results directly into the WMMA source tuples.
// Optional: bf16 store (Cbf), pattern mask (pm: where 0 -> -10000).
// Requires M%16==0, N%64==0, K%128==0.
// ---------------------------------------------------------------------------
__global__ void __launch_bounds__(256) gemm_bf16_kernel(
    const unsigned short* __restrict__ A, int lda, long long sA,
    const unsigned short* __restrict__ W, int ldw, long long sW,
    float* __restrict__ Cf, unsigned short* __restrict__ Cbf, int ldc, long long sC,
    int M, int N, int K,
    const float* __restrict__ bias, const float* __restrict__ bias2, float scale,
    const unsigned char* __restrict__ pm) {
  int batch = blockIdx.y;
  A += (size_t)batch * sA;
  W += (size_t)batch * sW;
  size_t cOff = (size_t)batch * sC;

  int wave = threadIdx.x >> 5;
  int lane = threadIdx.x & 31;
  int tilesN = N >> 6;
  int tile = blockIdx.x * 8 + wave;
  int tM = tile / tilesN;
  int tN = tile - tM * tilesN;
  if (tM * 16 >= M) return;

  const unsigned short* Arow = A + (size_t)tM * 16 * lda;
  const unsigned short* W0 = W + (size_t)(tN * 64) * ldw;

  v8f acc[4];
#pragma unroll
  for (int j = 0; j < 4; j++) acc[j] = vzero();

  // prologue: prime 4 fragment buffers (k = 0, 32, 64, 96)
  v16bf aB[4];
  v16bf bB[4][4];
#pragma unroll
  for (int u = 0; u < 4; u++) {
    aB[u] = load_a_frag(Arow, lda, 32 * u);
#pragma unroll
    for (int j = 0; j < 4; j++)
      bB[u][j] = load_b_frag(W0 + (size_t)(j * 16) * ldw, ldw, 32 * u);
  }

  for (int k0 = 128; k0 < K; k0 += 128) {
#pragma unroll
    for (int u = 0; u < 4; u++) {
#pragma unroll
      for (int j = 0; j < 4; j++) acc[j] = wmma_bf16(aB[u], bB[u][j], acc[j]);
      aB[u] = load_a_frag(Arow, lda, k0 + 32 * u);
#pragma unroll
      for (int j = 0; j < 4; j++)
        bB[u][j] = load_b_frag(W0 + (size_t)(j * 16) * ldw, ldw, k0 + 32 * u);
    }
  }
#pragma unroll
  for (int u = 0; u < 4; u++) {
#pragma unroll
    for (int j = 0; j < 4; j++) acc[j] = wmma_bf16(aB[u], bB[u][j], acc[j]);
  }

  int g = lane >> 4, nl = lane & 15;
#pragma unroll
  for (int j = 0; j < 4; j++) {
    int n = tN * 64 + j * 16 + nl;
    float bv = 0.f;
    if (bias) bv += bias[n];
    if (bias2) bv += bias2[n];
#pragma unroll
    for (int r = 0; r < 8; r++) {
      int m = tM * 16 + r + 8 * g;
      float v = acc[j][r] * scale + bv;
      size_t idx = cOff + (size_t)m * ldc + n;
      if (pm && !pm[idx]) v = -10000.f;
      if (Cbf) Cbf[idx] = f2bf(v);
      else     Cf[idx] = v;
    }
  }
}

// ---------------------------------------------------------------------------
// fp32 -> bf16 array convert (grid-stride)
// ---------------------------------------------------------------------------
__global__ void __launch_bounds__(256) cvt_kernel(const float* __restrict__ in,
                                                  unsigned short* __restrict__ out,
                                                  size_t n) {
  size_t i = (size_t)blockIdx.x * blockDim.x + threadIdx.x;
  size_t stride = (size_t)gridDim.x * blockDim.x;
  for (; i < n; i += stride) out[i] = f2bf(in[i]);
}

// ---------------------------------------------------------------------------
// Embedding: x[b,t,h] = tok_emb[ids[b,t],h] + pos_emb[t,h]  -> bf16
// ---------------------------------------------------------------------------
__global__ void __launch_bounds__(256) embed_kernel(const int* __restrict__ ids,
                                                    const float* __restrict__ tok,
                                                    const float* __restrict__ pos,
                                                    unsigned short* __restrict__ xbf) {
  size_t i = (size_t)blockIdx.x * 256 + threadIdx.x;
  if (i >= (size_t)DB * DT * DH) return;
  int row = (int)(i >> 10);  // b*T + t
  int h = (int)(i & (DH - 1));
  int t = row & (DT - 1);
  int id = ids[row];
  xbf[i] = f2bf(tok[(size_t)id * DH + h] + pos[(size_t)t * DH + h]);
}

// ---------------------------------------------------------------------------
// Persistent LSTM: 16 workgroups x 256 threads. WG k owns hidden units
// [64k, 64k+64), i.e. gate columns {c, 1024+c, 2048+c, 3072+c}. Per step:
// 16x256x1024 bf16 WMMA GEMM slice of h@w_hh^T (4-deep pipelined),
// pointwise update, ping-pong h buffer, per-step global arrival counter.
// ---------------------------------------------------------------------------
__global__ void __launch_bounds__(256) lstm_kernel(
    const float* __restrict__ gates_x,          // (B*T, 4096) includes b_ih+b_hh
    const unsigned short* __restrict__ whhbf,   // (4096, 1024) bf16
    unsigned short* __restrict__ hbf,           // 2 x (16 x 1024) bf16 (zeroed)
    float* __restrict__ cbuf,                   // (16 x 1024) f32 (zeroed)
    unsigned short* __restrict__ cat,           // (B*T, 2048): h goes to cols [0,1024)
    unsigned int* __restrict__ ctr) {           // 128 counters (zeroed)
  const int k = blockIdx.x;          // 0..15
  const int tid = threadIdx.x;
  const int wave = tid >> 5;
  const int lane = tid & 31;
  __shared__ float gbuf[16 * 256];   // local gate slice (only rows 0..7 used)

  const int lc0 = wave * 32;
  const int lc1 = wave * 32 + 16;
  const int gr0 = (lc0 >> 6) * DH + k * 64 + (lc0 & 63);
  const int gr1 = (lc1 >> 6) * DH + k * 64 + (lc1 & 63);
  const unsigned short* W0 = whhbf + (size_t)gr0 * DH;
  const unsigned short* W1 = whhbf + (size_t)gr1 * DH;

  for (int t = 0; t < DT; t++) {
    const unsigned short* hread = hbf + (size_t)(t & 1) * 16 * DH;
    v8f acc0 = vzero(), acc1 = vzero();

    v16bf aB[4], pB[4], qB[4];
#pragma unroll
    for (int u = 0; u < 4; u++) {
      aB[u] = load_a_frag(hread, DH, 32 * u);
      pB[u] = load_b_frag(W0, DH, 32 * u);
      qB[u] = load_b_frag(W1, DH, 32 * u);
    }
    for (int k0 = 128; k0 < DH; k0 += 128) {
#pragma unroll
      for (int u = 0; u < 4; u++) {
        acc0 = wmma_bf16(aB[u], pB[u], acc0);
        acc1 = wmma_bf16(aB[u], qB[u], acc1);
        aB[u] = load_a_frag(hread, DH, k0 + 32 * u);
        pB[u] = load_b_frag(W0, DH, k0 + 32 * u);
        qB[u] = load_b_frag(W1, DH, k0 + 32 * u);
      }
    }
#pragma unroll
    for (int u = 0; u < 4; u++) {
      acc0 = wmma_bf16(aB[u], pB[u], acc0);
      acc1 = wmma_bf16(aB[u], qB[u], acc1);
    }

    int g = lane >> 4, nl = lane & 15;
#pragma unroll
    for (int r = 0; r < 8; r++) {
      int m = r + 8 * g;
      gbuf[m * 256 + lc0 + nl] = acc0[r];
      gbuf[m * 256 + lc1 + nl] = acc1[r];
    }
    __syncthreads();

    for (int it = tid; it < 512; it += 256) {   // 8 batches x 64 units
      int bb = it >> 6, u = it & 63;
      int uu = k * 64 + u;
      const float* gx = gates_x + ((size_t)bb * DT + t) * (4 * DH);
      float gi = gx[uu]            + gbuf[bb * 256 + u];
      float gf = gx[DH + uu]       + gbuf[bb * 256 + 64 + u];
      float gg = gx[2 * DH + uu]   + gbuf[bb * 256 + 128 + u];
      float go = gx[3 * DH + uu]   + gbuf[bb * 256 + 192 + u];
      float si = 1.f / (1.f + expf(-gi));
      float sf = 1.f / (1.f + expf(-gf));
      float so = 1.f / (1.f + expf(-go));
      float cv = sf * cbuf[bb * DH + uu] + si * tanhf(gg);
      cbuf[bb * DH + uu] = cv;
      float hv = so * tanhf(cv);
      unsigned short hb = f2bf(hv);
      hbf[(size_t)((t + 1) & 1) * 16 * DH + bb * DH + uu] = hb;  // next-step h
      cat[((size_t)bb * DT + t) * (2 * DH) + uu] = hb;           // lstm_out
    }
    __syncthreads();

    // grid-wide barrier: all 16 WGs must publish h before next step reads it
    if (tid == 0) {
      __threadfence();
      atomicAdd(&ctr[t], 1u);
      while (__hip_atomic_load(&ctr[t], __ATOMIC_ACQUIRE, __HIP_MEMORY_SCOPE_AGENT) < 16u) {
        __builtin_amdgcn_s_sleep(2);
      }
    }
    __syncthreads();
  }
}

// ---------------------------------------------------------------------------
// LayerNorm over H=1024, write bf16; optionally also write transposed bf16
// (for kn -> knT, layout (b, h, s)).
// ---------------------------------------------------------------------------
__global__ void __launch_bounds__(256) ln_kernel(const float* __restrict__ in,
                                                 unsigned short* __restrict__ out,
                                                 unsigned short* __restrict__ outT,
                                                 const float* __restrict__ gam,
                                                 const float* __restrict__ bet,
                                                 int rowsPerBatch) {
  int row = blockIdx.x;
  int tid = threadIdx.x;
  const float* x = in + (size_t)row * DH;
  __shared__ float red[256];

  float vals[4];
  float s = 0.f;
#pragma unroll
  for (int i = 0; i < 4; i++) { vals[i] = x[tid + 256 * i]; s += vals[i]; }
  red[tid] = s; __syncthreads();
  for (int st = 128; st > 0; st >>= 1) { if (tid < st) red[tid] += red[tid + st]; __syncthreads(); }
  float mean = red[0] * (1.f / DH);
  __syncthreads();

  float v = 0.f;
#pragma unroll
  for (int i = 0; i < 4; i++) { float d = vals[i] - mean; v += d * d; }
  red[tid] = v; __syncthreads();
  for (int st = 128; st > 0; st >>= 1) { if (tid < st) red[tid] += red[tid + st]; __syncthreads(); }
  float inv = rsqrtf(red[0] * (1.f / DH) + 1e-5f);

#pragma unroll
  for (int i = 0; i < 4; i++) {
    int h = tid + 256 * i;
    float y = (vals[i] - mean) * inv * gam[h] + bet[h];
    unsigned short ub = f2bf(y);
    out[(size_t)row * DH + h] = ub;
    if (outT) {
      int b = row / rowsPerBatch, sr = row - b * rowsPerBatch;
      outT[((size_t)b * DH + h) * rowsPerBatch + sr] = ub;
    }
  }
}

// ---------------------------------------------------------------------------
// Masked softmax over S=256 per (b,t) row, matching reference exactly.
// ---------------------------------------------------------------------------
__global__ void __launch_bounds__(256) softmax_kernel(const float* __restrict__ scores,
                                                      unsigned short* __restrict__ probs,
                                                      const unsigned char* __restrict__ amask) {
  int row = blockIdx.x;        // b*T + t
  int tid = threadIdx.x;       // s
  int b = row >> 7;            // T = 128
  float v = scores[(size_t)row * DSS + tid];
  bool ok = amask[b * DSS + tid] != 0;
  float sv = ok ? v : -10000.f;

  __shared__ float red[256];
  red[tid] = sv; __syncthreads();
  for (int st = 128; st > 0; st >>= 1) { if (tid < st) red[tid] = fmaxf(red[tid], red[tid + st]); __syncthreads(); }
  float mx = red[0]; __syncthreads();

  float ex = ok ? expf(sv - mx) : 0.f;
  red[tid] = ex; __syncthreads();
  for (int st = 128; st > 0; st >>= 1) { if (tid < st) red[tid] += red[tid + st]; __syncthreads(); }
  float inv = 1.f / (red[0] + 1e-6f);
  probs[(size_t)row * DSS + tid] = f2bf(ex * inv);
}

// ---------------------------------------------------------------------------
extern "C" void kernel_launch(void* const* d_in, const int* in_sizes, int n_in,
                              void* d_out, int out_size, void* d_ws, size_t ws_size,
                              hipStream_t stream) {
  (void)in_sizes; (void)n_in; (void)out_size; (void)ws_size;
  const int* ids            = (const int*)d_in[0];
  const float* enc          = (const float*)d_in[1];
  const unsigned char* pmask = (const unsigned char*)d_in[2];
  const unsigned char* amask = (const unsigned char*)d_in[3];
  const float* tok  = (const float*)d_in[4];
  const float* pos  = (const float*)d_in[5];
  const float* w_ih = (const float*)d_in[6];
  const float* w_hh = (const float*)d_in[7];
  const float* b_ih = (const float*)d_in[8];
  const float* b_hh = (const float*)d_in[9];
  const float* wq   = (const float*)d_in[10];
  const float* bq   = (const float*)d_in[11];
  const float* wk   = (const float*)d_in[12];
  const float* bk   = (const float*)d_in[13];
  const float* wc   = (const float*)d_in[14];
  const float* bc   = (const float*)d_in[15];
  const float* wo   = (const float*)d_in[16];
  const float* bo   = (const float*)d_in[17];
  const float* g1   = (const float*)d_in[18];
  const float* be1  = (const float*)d_in[19];
  const float* g2   = (const float*)d_in[20];
  const float* be2  = (const float*)d_in[21];
  float* out = (float*)d_out;

  // ---- workspace layout (256B aligned) ----
  char* ws = (char*)d_ws;
  size_t off = 0;
  auto alloc = [&](size_t bytes) -> char* {
    char* p = ws + off;
    off = (off + bytes + 255) & ~(size_t)255;
    return p;
  };
  unsigned short* xbf   = (unsigned short*)alloc((size_t)DB * DT * DH * 2);
  unsigned short* encbf = (unsigned short*)alloc((size_t)DB * DSS * DH * 2);
  unsigned short* wihbf = (unsigned short*)alloc((size_t)4 * DH * DH * 2);
  unsigned short* whhbf = (unsigned short*)alloc((size_t)4 * DH * DH * 2);
  unsigned short* wqbf  = (unsigned short*)alloc((size_t)DH * DH * 2);
  unsigned short* wkbf  = (unsigned short*)alloc((size_t)DH * DH * 2);
  unsigned short* wcbf  = (unsigned short*)alloc((size_t)DH * 2 * DH * 2);
  unsigned short* wobf  = (unsigned short*)alloc((size_t)DV * DH * 2);
  float* gx             = (float*)alloc((size_t)DB * DT * 4 * DH * 4);
  unsigned short* cat   = (unsigned short*)alloc((size_t)DB * DT * 2 * DH * 2);
  char* state           = alloc(65536 + 65536 + 1024);  // hbf | cbuf | ctr
  unsigned short* hbf   = (unsigned short*)state;
  float* cbuf           = (float*)(state + 65536);
  unsigned int* ctr     = (unsigned int*)(state + 131072);
  float* qraw           = (float*)alloc((size_t)DB * DT * DH * 4);
  float* kraw           = (float*)alloc((size_t)DB * DSS * DH * 4);
  unsigned short* qnbf  = (unsigned short*)alloc((size_t)DB * DT * DH * 2);
  unsigned short* knbf  = (unsigned short*)alloc((size_t)DB * DSS * DH * 2);
  unsigned short* knT   = (unsigned short*)alloc((size_t)DB * DH * DSS * 2);
  float* scores         = (float*)alloc((size_t)DB * DT * DSS * 4);
  unsigned short* attnw = (unsigned short*)alloc((size_t)DB * DT * DSS * 2);
  float* hidraw         = (float*)alloc((size_t)DB * DT * DH * 4);
  unsigned short* hidbf = (unsigned short*)alloc((size_t)DB * DT * DH * 2);

  hipMemsetAsync(state, 0, 65536 + 65536 + 1024, stream);  // h0, c0, barriers

  auto cvt = [&](const float* in, unsigned short* o, size_t n) {
    cvt_kernel<<<dim3(4096), 256, 0, stream>>>(in, o, n);
  };
  cvt(w_ih, wihbf, (size_t)4 * DH * DH);
  cvt(w_hh, whhbf, (size_t)4 * DH * DH);
  cvt(wq, wqbf, (size_t)DH * DH);
  cvt(wk, wkbf, (size_t)DH * DH);
  cvt(wc, wcbf, (size_t)DH * 2 * DH);
  cvt(wo, wobf, (size_t)DV * DH);
  cvt(enc, encbf, (size_t)DB * DSS * DH);

  embed_kernel<<<dim3((DB * DT * DH) / 256), 256, 0, stream>>>(ids, tok, pos, xbf);

  auto gemm = [&](const unsigned short* A, int lda, long long sA,
                  const unsigned short* W, int ldw, long long sW,
                  float* Cf, unsigned short* Cbf, int ldc, long long sC,
                  int M, int N, int K, int batches,
                  const float* bias, const float* bias2, float scale,
                  const unsigned char* pm) {
    int tiles = (M / 16) * (N / 64);
    dim3 grid((tiles + 7) / 8, batches);
    gemm_bf16_kernel<<<grid, 256, 0, stream>>>(A, lda, sA, W, ldw, sW, Cf, Cbf,
                                               ldc, sC, M, N, K, bias, bias2,
                                               scale, pm);
  };

  // gates_x = x @ w_ih^T + (b_ih + b_hh)     (1024 x 4096 x 1024)
  gemm(xbf, DH, 0, wihbf, DH, 0, gx, nullptr, 4 * DH, 0,
       DB * DT, 4 * DH, DH, 1, b_ih, b_hh, 1.f, nullptr);

  // sequential LSTM (persistent, 16 WGs, per-step global barrier)
  lstm_kernel<<<dim3(16), 256, 0, stream>>>(gx, whhbf, hbf, cbuf, cat, ctr);

  // qraw = lstm_out @ wq^T + bq   (lstm_out lives in cat[:, 0:1024], lda=2048)
  gemm(cat, 2 * DH, 0, wqbf, DH, 0, qraw, nullptr, DH, 0,
       DB * DT, DH, DH, 1, bq, nullptr, 1.f, nullptr);
  ln_kernel<<<dim3(DB * DT), 256, 0, stream>>>(qraw, qnbf, nullptr, g1, be1, 1);

  // kraw = enc @ wk^T + bk        (2048 x 1024 x 1024)
  gemm(encbf, DH, 0, wkbf, DH, 0, kraw, nullptr, DH, 0,
       DB * DSS, DH, DH, 1, bk, nullptr, 1.f, nullptr);
  ln_kernel<<<dim3(DB * DSS), 256, 0, stream>>>(kraw, knbf, knT, g1, be1, DSS);

  // scores[b] = scale * qn[b] @ kn[b]^T      (batched 128 x 256 x 1024)
  gemm(qnbf, DH, (long long)DT * DH, knbf, DH, (long long)DSS * DH,
       scores, nullptr, DSS, (long long)DT * DSS,
       DT, DSS, DH, DB, nullptr, nullptr, 0.015625f /* H^-0.5 * 0.5 */, nullptr);

  softmax_kernel<<<dim3(DB * DT), 256, 0, stream>>>(scores, attnw, amask);

  // ctx[b] = probs[b] @ kn[b]  via knT, stored bf16 into cat[:, 1024:2048]
  gemm(attnw, DSS, (long long)DT * DSS, knT, DSS, (long long)DH * DSS,
       nullptr, cat + DH, 2 * DH, (long long)DT * 2 * DH,
       DT, DH, DSS, DB, nullptr, nullptr, 1.f, nullptr);

  // hidden_raw = cat @ wc^T + bc             (1024 x 1024 x 2048)
  gemm(cat, 2 * DH, 0, wcbf, 2 * DH, 0, hidraw, nullptr, DH, 0,
       DB * DT, DH, 2 * DH, 1, bc, nullptr, 1.f, nullptr);
  ln_kernel<<<dim3(DB * DT), 256, 0, stream>>>(hidraw, hidbf, nullptr, g2, be2, 1);

  // logits = hidden @ wo^T + bo, pattern-masked  (1024 x 32000 x 1024)
  gemm(hidbf, DH, 0, wobf, DH, 0, out, nullptr, DV, 0,
       DB * DT, DV, DH, 1, bo, nullptr, 1.f, pmask);
}
